// GAT_Classifier_LL_34815004901855
// MI455X (gfx1250) — matrix operand order repaired and checked by reading, not compiled, not run on hardware
//
#include <hip/hip_runtime.h>
#include <hip/hip_bf16.h>

typedef __attribute__((ext_vector_type(16))) _Float16     v16h;
typedef __attribute__((ext_vector_type(8)))  float        v8f;
typedef __attribute__((ext_vector_type(8)))  unsigned int v8u;

#define LRELU_SLOPE 0.2f
#define LN_EPS 1e-5f

union Frag16 {            // build a 16-half WMMA operand from two dword-quads
    uint4 q[2];
    v8u   u;
    v16h  h;
};

// ---------------------------------------------------------------------------
// float atomic max via int/uint ordering trick (valid with -inf init)
// ---------------------------------------------------------------------------
__device__ __forceinline__ void atomicMaxFloat(float* addr, float v) {
    if (v >= 0.0f) atomicMax((int*)addr, __float_as_int(v));
    else           atomicMin((unsigned int*)addr, __float_as_uint(v));
}

// ---------------------------------------------------------------------------
// f32 -> f16 conversion
// ---------------------------------------------------------------------------
__global__ void k_f32_to_f16(const float* __restrict__ X, _Float16* __restrict__ Y, int n) {
    int i = blockIdx.x * blockDim.x + threadIdx.x;
    if (i < n) Y[i] = (_Float16)X[i];
}

// ---------------------------------------------------------------------------
// Pack W [K, NOUT] f32 row-major into f16 K-pair words: Bpack[p*NOUT+n] holds
// (W[2p][n], W[2p+1][n]) in low/high halves (WMMA B-fragment element order).
// ---------------------------------------------------------------------------
__global__ void k_packW(const float* __restrict__ W, unsigned int* __restrict__ Bpack,
                        int K, int NOUT) {
    int i = blockIdx.x * blockDim.x + threadIdx.x;
    int total = (K >> 1) * NOUT;
    if (i >= total) return;
    int p = i / NOUT, n = i % NOUT;
    _Float16 lo = (_Float16)W[(2 * p) * NOUT + n];
    _Float16 hi = (_Float16)W[(2 * p + 1) * NOUT + n];
    unsigned int ulo = (unsigned int)(*(unsigned short*)&lo);
    unsigned int uhi = (unsigned int)(*(unsigned short*)&hi);
    Bpack[i] = ulo | (uhi << 16);
}

// ---------------------------------------------------------------------------
// WMMA GEMM, compile-time K/NOUT.
//  - W staged once per block into LDS, transposed to [col][k-pair] so each
//    lane's B fragment is two 4-dword runs 8 dwords apart (2x ds_load_b128).
//  - each wave owns a 16-row strip; A fragments loaded once (global b128),
//    reused across all NOUT/16 column tiles -> K/32 * NOUT/16 WMMAs per wave.
//  Fragment maps (ISA 7.12.2, wave32):
//    A (16x32 f16): lane m=lane&15, dword pair p = g*8 + half*4 + jj
//    B (32x16 f16): pair p = g*8 + half*4 + jj, col n = lane&15
//    C/D (16x16 f32): VGPR r -> row r + half*8, col = lane&15
// ---------------------------------------------------------------------------
template <int K, int NOUT>
__global__ void __launch_bounds__(256)
k_gemm_wmma(const _Float16* __restrict__ A16,
            const unsigned int* __restrict__ Bpack,
            float* __restrict__ Cout, int M) {
    constexpr int KP = K / 2;      // dword pairs along K
    constexpr int KB = K / 32;     // WMMA K-steps
    constexpr int TN = NOUT / 16;  // column tiles
    __shared__ __attribute__((aligned(32))) unsigned int Bs[NOUT * KP];

    // cooperative stage + transpose of W into LDS: Bs[col][pair]
    for (int idx = threadIdx.x; idx < KP * NOUT; idx += blockDim.x) {
        int p = idx / NOUT, col = idx % NOUT;
        Bs[col * KP + p] = Bpack[idx];
    }
    __syncthreads();

    int w    = threadIdx.x >> 5;
    int lane = threadIdx.x & 31;
    int l16  = lane & 15, hf = lane >> 4;
    int tm   = blockIdx.x * 8 + w;          // 16-row tile index
    if (tm * 16 >= M) return;               // wave-uniform exit

    // hoist all A fragments for this row strip
    const uint4* Arow = (const uint4*)(A16 + (size_t)(tm * 16 + l16) * K);
    Frag16 afrag[KB];
#pragma unroll
    for (int kb = 0; kb < KB; ++kb) {
        afrag[kb].q[0] = Arow[kb * 4 + hf];      // dwords kb*16 + hf*4 + 0..3
        afrag[kb].q[1] = Arow[kb * 4 + 2 + hf];  // dwords kb*16 + 8 + hf*4 + 0..3
    }

    float* Cbase = Cout + (size_t)(tm * 16 + hf * 8) * NOUT + l16;
#pragma unroll
    for (int tn = 0; tn < TN; ++tn) {
        v8f acc = {0.f, 0.f, 0.f, 0.f, 0.f, 0.f, 0.f, 0.f};
        const unsigned int* bcol = &Bs[(tn * 16 + l16) * KP + hf * 4];
#pragma unroll
        for (int kb = 0; kb < KB; ++kb) {
            Frag16 b;
            b.q[0] = *(const uint4*)(bcol + kb * 16);
            b.q[1] = *(const uint4*)(bcol + kb * 16 + 8);
            acc = __builtin_amdgcn_wmma_f32_16x16x32_f16(
                false, afrag[kb].h, false, b.h, (short)0, acc, false, false);
        }
#pragma unroll
        for (int r = 0; r < 8; ++r)
            Cbase[tn * 16 + (size_t)r * NOUT] = acc[r];
    }
}

// ---------------------------------------------------------------------------
// Per-node attention logits + init of m / denom / agg
// ---------------------------------------------------------------------------
template <int H, int C>
__global__ void k_node_prep(const float* __restrict__ h,
                            const float* __restrict__ a_src, const float* __restrict__ a_dst,
                            float* __restrict__ esrc, float* __restrict__ edst,
                            float* __restrict__ m, float* __restrict__ denom,
                            float* __restrict__ agg, int Nn) {
    constexpr int D = H * C;
    int n = blockIdx.x * blockDim.x + threadIdx.x;
    if (n >= Nn) return;
    const float4* hr = (const float4*)(h + (size_t)n * D);
    const float4* as4 = (const float4*)a_src;
    const float4* ad4 = (const float4*)a_dst;
#pragma unroll
    for (int hd = 0; hd < H; ++hd) {
        float s = 0.f, d = 0.f;
#pragma unroll
        for (int c4 = 0; c4 < C / 4; ++c4) {
            float4 v = hr[hd * (C / 4) + c4];
            float4 a = as4[hd * (C / 4) + c4];
            float4 b = ad4[hd * (C / 4) + c4];
            s += v.x * a.x + v.y * a.y + v.z * a.z + v.w * a.w;
            d += v.x * b.x + v.y * b.y + v.z * b.z + v.w * b.w;
        }
        esrc[n * H + hd] = s;
        edst[n * H + hd] = d;
        ((unsigned int*)m)[n * H + hd] = 0xFF800000u;  // -inf
        denom[n * H + hd] = 0.f;
    }
    float4* ar = (float4*)(agg + (size_t)n * D);
    float4 z = {0.f, 0.f, 0.f, 0.f};
#pragma unroll
    for (int i = 0; i < D / 4; ++i) ar[i] = z;
}

// ---------------------------------------------------------------------------
// Edge pass 1: segment max of leaky-relu logits into m[dst,h]
// ---------------------------------------------------------------------------
template <int H>
__global__ void k_edge_max(const int* __restrict__ srcI, const int* __restrict__ dstI,
                           int E, int Nn,
                           const float* __restrict__ esrc, const float* __restrict__ edst,
                           float* __restrict__ m) {
    int t = blockIdx.x * blockDim.x + threadIdx.x;
    int total = (E + Nn) * H;
    if (t >= total) return;
    int hd = t % H;
    int e  = t / H;
    int s = (e < E) ? srcI[e] : (e - E);
    int d = (e < E) ? dstI[e] : (e - E);
    float x = esrc[s * H + hd] + edst[d * H + hd];
    x = (x > 0.f) ? x : LRELU_SLOPE * x;
    atomicMaxFloat(&m[d * H + hd], x);
}

// ---------------------------------------------------------------------------
// Edge pass 2: w = exp(e - m[dst]); denom[dst] += w; agg[dst] += w * h[src]
// ---------------------------------------------------------------------------
template <int H, int C>
__global__ void k_edge_acc(const int* __restrict__ srcI, const int* __restrict__ dstI,
                           int E, int Nn,
                           const float* __restrict__ esrc, const float* __restrict__ edst,
                           const float* __restrict__ m, float* __restrict__ denom,
                           float* __restrict__ agg, const float* __restrict__ h) {
    int t = blockIdx.x * blockDim.x + threadIdx.x;
    int total = (E + Nn) * H;
    if (t >= total) return;
    int hd = t % H;
    int e  = t / H;
    int s = (e < E) ? srcI[e] : (e - E);
    int d = (e < E) ? dstI[e] : (e - E);
    float x = esrc[s * H + hd] + edst[d * H + hd];
    x = (x > 0.f) ? x : LRELU_SLOPE * x;
    float mv = m[d * H + hd];
    if (!(mv >= -3.0e38f)) mv = 0.f;  // -inf / NaN -> 0 (matches reference)
    float w = expf(x - mv);
    atomicAdd(&denom[d * H + hd], w);
    const float4* hs = (const float4*)(h + (size_t)s * H * C + hd * C);
    float* ag = agg + (size_t)d * H * C + hd * C;
#pragma unroll
    for (int c4 = 0; c4 < C / 4; ++c4) {
        float4 v = hs[c4];
        atomicAdd(&ag[c4 * 4 + 0], w * v.x);
        atomicAdd(&ag[c4 * 4 + 1], w * v.y);
        atomicAdd(&ag[c4 * 4 + 2], w * v.z);
        atomicAdd(&ag[c4 * 4 + 3], w * v.w);
    }
}

// ---------------------------------------------------------------------------
// Node epilogue: out = agg/denom + bias -> LayerNorm -> ELU
// ---------------------------------------------------------------------------
template <int H, int C>
__global__ void k_norm_ln_elu(const float* __restrict__ agg, const float* __restrict__ denom,
                              const float* __restrict__ bias,
                              const float* __restrict__ gamma, const float* __restrict__ beta,
                              _Float16* __restrict__ out16, float* __restrict__ out32,
                              int Nn) {
    constexpr int D = H * C;
    int n = blockIdx.x * blockDim.x + threadIdx.x;
    if (n >= Nn) return;
    float v[D];
    float idn[H];
#pragma unroll
    for (int hd = 0; hd < H; ++hd) idn[hd] = 1.0f / (denom[n * H + hd] + 1e-16f);
    float mu = 0.f;
#pragma unroll
    for (int i = 0; i < D; ++i) {
        float x = agg[(size_t)n * D + i] * idn[i / C] + bias[i];
        v[i] = x;
        mu += x;
    }
    mu *= (1.0f / D);
    float var = 0.f;
#pragma unroll
    for (int i = 0; i < D; ++i) { float dx = v[i] - mu; var += dx * dx; }
    var *= (1.0f / D);
    float inv = rsqrtf(var + LN_EPS);
#pragma unroll
    for (int i = 0; i < D; ++i) {
        float y = (v[i] - mu) * inv * gamma[i] + beta[i];
        v[i] = (y > 0.f) ? y : (expf(y) - 1.f);  // ELU
    }
    if (out16) {
#pragma unroll
        for (int i = 0; i < D; ++i) out16[(size_t)n * D + i] = (_Float16)v[i];
    }
    if (out32) {
#pragma unroll
        for (int i = 0; i < D; ++i) out32[(size_t)n * D + i] = v[i];
    }
}

// ---------------------------------------------------------------------------
// Graph pooling
// ---------------------------------------------------------------------------
__global__ void k_pool_init(float* __restrict__ cnt, float* __restrict__ psum,
                            unsigned int* __restrict__ pmaxb, int G, int C) {
    int i = blockIdx.x * blockDim.x + threadIdx.x;
    if (i < G) cnt[i] = 0.f;
    if (i < G * C) { psum[i] = 0.f; pmaxb[i] = 0xFF800000u; }
}

__global__ void k_pool_acc(const float* __restrict__ h3, const int* __restrict__ batch,
                           float* __restrict__ cnt, float* __restrict__ psum,
                           float* __restrict__ pmax, int Nn) {
    constexpr int C = 32;
    int n = blockIdx.x * blockDim.x + threadIdx.x;
    if (n >= Nn) return;
    int g = batch[n];
    atomicAdd(&cnt[g], 1.0f);
    const float4* hr = (const float4*)(h3 + (size_t)n * C);
    float* ps = psum + g * C;
    float* pm = pmax + g * C;
#pragma unroll
    for (int c4 = 0; c4 < C / 4; ++c4) {
        float4 v = hr[c4];
        atomicAdd(&ps[c4 * 4 + 0], v.x);
        atomicAdd(&ps[c4 * 4 + 1], v.y);
        atomicAdd(&ps[c4 * 4 + 2], v.z);
        atomicAdd(&ps[c4 * 4 + 3], v.w);
        atomicMaxFloat(&pm[c4 * 4 + 0], v.x);
        atomicMaxFloat(&pm[c4 * 4 + 1], v.y);
        atomicMaxFloat(&pm[c4 * 4 + 2], v.z);
        atomicMaxFloat(&pm[c4 * 4 + 3], v.w);
    }
}

// ---------------------------------------------------------------------------
// Per-graph classifier MLP: [mean|max](64) -> 32 -> 16 -> 1
// ---------------------------------------------------------------------------
__global__ void k_mlp(const float* __restrict__ cnt, const float* __restrict__ psum,
                      const float* __restrict__ pmax,
                      const float* __restrict__ cW1, const float* __restrict__ cb1,
                      const float* __restrict__ cW2, const float* __restrict__ cb2,
                      const float* __restrict__ cW3, const float* __restrict__ cb3,
                      float* __restrict__ out, int G) {
    constexpr int C = 32;
    int g = blockIdx.x * blockDim.x + threadIdx.x;
    if (g >= G) return;
    float f[64];
    float ic = 1.0f / fmaxf(cnt[g], 1.0f);
    for (int j = 0; j < C; ++j) {
        f[j]     = psum[g * C + j] * ic;
        f[C + j] = pmax[g * C + j];
    }
    float h1[32];
    for (int i = 0; i < 32; ++i) {
        float s = cb1[i];
        for (int j = 0; j < 64; ++j) s += f[j] * cW1[i * 64 + j];
        h1[i] = fmaxf(s, 0.f);
    }
    float h2[16];
    for (int i = 0; i < 16; ++i) {
        float s = cb2[i];
        for (int j = 0; j < 32; ++j) s += h1[j] * cW2[i * 32 + j];
        h2[i] = fmaxf(s, 0.f);
    }
    float s = cb3[0];
    for (int j = 0; j < 16; ++j) s += h2[j] * cW3[j];
    out[g] = s;
}

// ---------------------------------------------------------------------------
// Host launch
// ---------------------------------------------------------------------------
extern "C" void kernel_launch(void* const* d_in, const int* in_sizes, int n_in,
                              void* d_out, int out_size, void* d_ws, size_t ws_size,
                              hipStream_t stream) {
    const float* x   = (const float*)d_in[0];
    const int*   ei  = (const int*)d_in[1];
    const int*   bat = (const int*)d_in[2];
    const float* W1  = (const float*)d_in[4];
    const float* as1 = (const float*)d_in[5];
    const float* ad1 = (const float*)d_in[6];
    const float* b1  = (const float*)d_in[7];
    const float* g1  = (const float*)d_in[8];
    const float* be1 = (const float*)d_in[9];
    const float* W2  = (const float*)d_in[10];
    const float* as2 = (const float*)d_in[11];
    const float* ad2 = (const float*)d_in[12];
    const float* b2  = (const float*)d_in[13];
    const float* g2  = (const float*)d_in[14];
    const float* be2 = (const float*)d_in[15];
    const float* W3  = (const float*)d_in[16];
    const float* as3 = (const float*)d_in[17];
    const float* ad3 = (const float*)d_in[18];
    const float* b3  = (const float*)d_in[19];
    const float* g3  = (const float*)d_in[20];
    const float* be3 = (const float*)d_in[21];
    const float* cW1 = (const float*)d_in[22];
    const float* cb1 = (const float*)d_in[23];
    const float* cW2 = (const float*)d_in[24];
    const float* cb2 = (const float*)d_in[25];
    const float* cW3 = (const float*)d_in[26];
    const float* cb3 = (const float*)d_in[27];

    const int N  = in_sizes[2];      // one batch entry per node
    const int E  = in_sizes[1] / 2;
    const int G  = out_size;
    const int F0 = in_sizes[0] / N;  // 32

    auto al = [](size_t o) { return (o + 255) & ~(size_t)255; };
    char* ws = (char*)d_ws;
    size_t off = 0;
    _Float16* Xh = (_Float16*)(ws + off); off = al(off + (size_t)N * 128 * sizeof(_Float16));
    float* h     = (float*)(ws + off);    off = al(off + (size_t)N * 128 * sizeof(float));
    float* agg   = (float*)(ws + off);    off = al(off + (size_t)N * 128 * sizeof(float));
    float* esrc  = (float*)(ws + off);    off = al(off + (size_t)N * 4 * sizeof(float));
    float* edst  = (float*)(ws + off);    off = al(off + (size_t)N * 4 * sizeof(float));
    float* mbuf  = (float*)(ws + off);    off = al(off + (size_t)N * 4 * sizeof(float));
    float* den   = (float*)(ws + off);    off = al(off + (size_t)N * 4 * sizeof(float));
    unsigned int* Bpack = (unsigned int*)(ws + off); off = al(off + (size_t)64 * 128 * 4);
    float* cnt  = (float*)(ws + off);     off = al(off + (size_t)G * sizeof(float));
    float* psum = (float*)(ws + off);     off = al(off + (size_t)G * 32 * sizeof(float));
    float* pmax = (float*)(ws + off);     off = al(off + (size_t)G * 32 * sizeof(float));
    (void)ws_size; (void)n_in;

    const int T = 256;
    const int* srcI = ei;
    const int* dstI = ei + E;
    const int gemmBlocks = (N / 16 + 7) / 8;  // 8 row-tiles per block

#define RUN_GAT(Kc, NOUTc, Hc, Cc, W, ASR, ADR, BIAS, GAM, BET, OUT16, OUT32)                     \
    do {                                                                                          \
        int pairs = ((Kc) / 2) * (NOUTc);                                                         \
        k_packW<<<(pairs + T - 1) / T, T, 0, stream>>>(W, Bpack, (Kc), (NOUTc));                  \
        k_gemm_wmma<Kc, NOUTc><<<gemmBlocks, 256, 0, stream>>>(Xh, Bpack, h, N);                  \
        k_node_prep<Hc, Cc><<<(N + T - 1) / T, T, 0, stream>>>(h, ASR, ADR, esrc, edst, mbuf,     \
                                                               den, agg, N);                      \
        int tE = (E + N) * (Hc);                                                                  \
        k_edge_max<Hc><<<(tE + T - 1) / T, T, 0, stream>>>(srcI, dstI, E, N, esrc, edst, mbuf);   \
        k_edge_acc<Hc, Cc><<<(tE + T - 1) / T, T, 0, stream>>>(srcI, dstI, E, N, esrc, edst,      \
                                                               mbuf, den, agg, h);                \
        k_norm_ln_elu<Hc, Cc><<<(N + T - 1) / T, T, 0, stream>>>(agg, den, BIAS, GAM, BET,        \
                                                                 OUT16, OUT32, N);                \
    } while (0)

    // layer 1: f32 x -> f16, GAT(32 -> 4x32 concat)
    k_f32_to_f16<<<(N * F0 + T - 1) / T, T, 0, stream>>>(x, Xh, N * F0);
    RUN_GAT(32, 128, 4, 32, W1, as1, ad1, b1, g1, be1, Xh, (float*)nullptr);
    // layer 2: GAT(128 -> 4x32 concat)
    RUN_GAT(128, 128, 4, 32, W2, as2, ad2, b2, g2, be2, Xh, (float*)nullptr);
    // layer 3: GAT(128 -> 1x32 mean); f32 output reuses h buffer
    RUN_GAT(128, 32, 1, 32, W3, as3, ad3, b3, g3, be3, (_Float16*)nullptr, h);

    // pooling + classifier
    k_pool_init<<<(G * 32 + T - 1) / T, T, 0, stream>>>(cnt, psum, (unsigned int*)pmax, G, 32);
    k_pool_acc<<<(N + T - 1) / T, T, 0, stream>>>(h, bat, cnt, psum, pmax, N);
    k_mlp<<<(G + 63) / 64, 64, 0, stream>>>(cnt, psum, pmax, cW1, cb1, cW2, cb2, cW3, cb3,
                                            (float*)d_out, G);
#undef RUN_GAT
}
